// YawNetVLAD_13280038880029
// MI455X (gfx1250) — matrix operand order repaired and checked by reading, not compiled, not run on hardware
//
#include <hip/hip_runtime.h>
#include <hip/hip_bf16.h>

typedef __attribute__((ext_vector_type(16))) _Float16 v16h;
typedef __attribute__((ext_vector_type(8)))  float    v8f;

#define WMMA_F16(A,B,C) __builtin_amdgcn_wmma_f32_16x16x32_f16(false,(A),false,(B),(short)0,(C),false,false)

static constexpr int NB   = 8;
static constexpr int CDIM = 128;
static constexpr int KCL  = 64;
static constexpr int HDIM = 16;
static constexpr int WDIM = 900;
static constexpr int HW   = HDIM * WDIM;       // 14400
static constexpr int SEG  = 10;
static constexpr int NSEG = 90;
static constexpr int NWIN = 30;
static constexpr int OUTD = 256;
static constexpr int DCON = HDIM * SEG;        // 160 contraction per segment
static constexpr int TILE = 512;               // halves per 16x32 / 32x16 fragment tile

// Fragment-native swizzles: value (m,kk) of an A tile (16x32, f16) lives at the
// LDS half-index the WMMA A operand expects for its (lane, element); same for B.
// A: lane = m + 16*((kk>>3)&1), elem = ((kk>>4)<<3) + (kk&7)
__device__ __forceinline__ int a_swz(int m, int kk) {
    return ((m + (((kk >> 3) & 1) << 4)) << 4) + (((kk >> 4) << 3) + (kk & 7));
}
// B: lane = n + 16*((k>>3)&1), elem = ((k>>4)<<3) + (k&7)
__device__ __forceinline__ int b_swz(int k, int n) {
    return ((n + (((k >> 3) & 1) << 4)) << 4) + (((k >> 4) << 3) + (k & 7));
}
__device__ __forceinline__ v16h load_frag(const _Float16* tile, int lane) {
    return *(const v16h*)(tile + (lane << 4));   // 32B contiguous per lane -> ds_load_b128 x2
}

// ---------------------------------------------------------------------------
// Kernel 1: 1x1 conv (WMMA) + softmax over clusters -> soft assignments (f16)
// grid: NB * (HW/64), block: 128 (4 waves, 16 positions each)
// ---------------------------------------------------------------------------
__global__ __launch_bounds__(128) void k1_assign(const float* __restrict__ x,
                                                 const float* __restrict__ conv_w,
                                                 const float* __restrict__ conv_b,
                                                 _Float16* __restrict__ a_out) {
    __shared__ __align__(32) _Float16 sX[4 * 4 * TILE];  // A tiles [mt][ch], 16 KB
    __shared__ __align__(32) _Float16 sW[4 * 4 * TILE];  // B tiles [kt][ch], 16 KB
    __shared__ float    sL[64][KCL + 1];                 // logits [pos][k]
    __shared__ float    sB[KCL];
    const int t  = threadIdx.x;
    const int n  = blockIdx.x / (HW / 64);
    const int p0 = (blockIdx.x % (HW / 64)) * 64;

    // conv_w (K x C, row-major) -> B tiles: col n = k&15, row = c within chunk
    for (int e = t; e < KCL * CDIM; e += 128) {
        const int k = e >> 7, c = e & 127;
        sW[((k >> 4) * 4 + (c >> 5)) * TILE + b_swz(c & 31, k & 15)] = (_Float16)conv_w[e];
    }
    if (t < KCL) sB[t] = conv_b[t];
    // x tile -> A tiles: row m = p&15, col = c within chunk
    const float* xn = x + (size_t)n * CDIM * HW;
    for (int e = t; e < CDIM * 64; e += 128) {
        const int c = e >> 6, p = e & 63;
        sX[((p >> 4) * 4 + (c >> 5)) * TILE + a_swz(p & 15, c & 31)] =
            (_Float16)xn[(size_t)c * HW + p0 + p];
    }
    __syncthreads();

    const int lane = t & 31, wave = t >> 5, hf = lane >> 4, l16 = lane & 15;
    const int m0 = wave * 16;
    for (int kt = 0; kt < 4; ++kt) {                // 4 cluster tiles of 16
        v8f acc = {};
        #pragma unroll
        for (int ch = 0; ch < 4; ++ch) {            // C = 4 chunks of 32
            const v16h af = load_frag(&sX[(wave * 4 + ch) * TILE], lane);
            const v16h bf = load_frag(&sW[(kt   * 4 + ch) * TILE], lane);
            acc = WMMA_F16(af, bf, acc);
        }
        #pragma unroll
        for (int r = 0; r < 8; ++r)
            sL[m0 + (hf << 3) + r][kt * 16 + l16] = acc[r];
    }
    __syncthreads();

    if (t < 64) {   // softmax over 64 clusters for one position
        float mx = -1e30f;
        for (int k = 0; k < KCL; ++k) mx = fmaxf(mx, sL[t][k] + sB[k]);
        float s = 0.f;
        for (int k = 0; k < KCL; ++k) s += __expf(sL[t][k] + sB[k] - mx);
        const float inv = 1.f / s;
        _Float16* ap = a_out + (size_t)n * KCL * HW + p0 + t;
        for (int k = 0; k < KCL; ++k)
            ap[(size_t)k * HW] = (_Float16)(__expf(sL[t][k] + sB[k] - mx) * inv);
    }
}

// ---------------------------------------------------------------------------
// Kernel 2: per-segment weighted sum wxseg[n][q][k][c] = sum_{h,w in seg} a*x
//           (WMMA, M=64 clusters, N=128 channels, K=160) + aseg[n][k][q]
// grid: NB*NSEG, block: 256 (8 waves: 4 k-tiles x 2 c-halves)
// ---------------------------------------------------------------------------
__global__ __launch_bounds__(256) void k2_wxseg(const float* __restrict__ x,
                                                const _Float16* __restrict__ a_in,
                                                float* __restrict__ wx_out,
                                                float* __restrict__ aseg_out) {
    __shared__ __align__(32) _Float16 sA[4 * 5 * TILE];   // A tiles [kt][ch], 20 KB
    __shared__ __align__(32) _Float16 sXl[8 * 5 * TILE];  // B tiles [ct][ch], 40 KB
    const int t  = threadIdx.x;
    const int n  = blockIdx.x / NSEG, q = blockIdx.x % NSEG;
    const int w0 = q * SEG;

    const _Float16* an = a_in + (size_t)n * KCL * HW;
    for (int e = t; e < KCL * DCON; e += 256) {
        const int k = e / DCON, d = e % DCON, h = d / SEG, wl = d % SEG;
        sA[((k >> 4) * 5 + (d >> 5)) * TILE + a_swz(k & 15, d & 31)] =
            an[(size_t)k * HW + h * WDIM + w0 + wl];
    }
    const float* xn = x + (size_t)n * CDIM * HW;
    for (int e = t; e < CDIM * DCON; e += 256) {
        const int c = e / DCON, d = e % DCON, h = d / SEG, wl = d % SEG;
        sXl[((c >> 4) * 5 + (d >> 5)) * TILE + b_swz(d & 31, c & 15)] =
            (_Float16)xn[(size_t)c * HW + h * WDIM + w0 + wl];
    }
    __syncthreads();

    if (t < KCL) {  // segment sum of assignments (swizzle-aware walk)
        float s = 0.f;
        const int kt = t >> 4, m = t & 15;
        for (int d = 0; d < DCON; ++d)
            s += (float)sA[(kt * 5 + (d >> 5)) * TILE + a_swz(m, d & 31)];
        aseg_out[((size_t)n * KCL + t) * NSEG + q] = s;
    }

    const int lane = t & 31, wave = t >> 5, hf = lane >> 4, l16 = lane & 15;
    const int kt = wave >> 1, ch2 = wave & 1;       // c half: tiles ch2*4 .. ch2*4+3
    v8f acc[4] = {};
    #pragma unroll
    for (int ch = 0; ch < 5; ++ch) {                // 160 = 5 chunks of 32
        const v16h af = load_frag(&sA[(kt * 5 + ch) * TILE], lane);
        #pragma unroll
        for (int ct = 0; ct < 4; ++ct) {
            const v16h bf = load_frag(&sXl[((ch2 * 4 + ct) * 5 + ch) * TILE], lane);
            acc[ct] = WMMA_F16(af, bf, acc[ct]);
        }
    }
    float* out = wx_out + (size_t)blockIdx.x * (KCL * CDIM);
    #pragma unroll
    for (int ct = 0; ct < 4; ++ct)
        #pragma unroll
        for (int r = 0; r < 8; ++r)
            out[(kt * 16 + (hf << 3) + r) * CDIM + (ch2 * 4 + ct) * 16 + l16] = acc[ct][r];
}

// ---------------------------------------------------------------------------
// Kernel 3: window = sum of 20 segments (circular, start seg (3j+80)%90),
//           subtract centers*awin, intra-norm per cluster, global norm,
//           emit f16 vlad row (8192 wide). grid: NB*NWIN, block 256.
// ---------------------------------------------------------------------------
__global__ __launch_bounds__(256) void k3_window_norm(const float* __restrict__ wx,
                                                      const float* __restrict__ aseg,
                                                      const float* __restrict__ centers,
                                                      _Float16* __restrict__ vlad) {
    __shared__ float vals[KCL * CDIM];   // 32 KB
    __shared__ float sAw[KCL];
    __shared__ float sNorm[KCL];
    __shared__ float red[256];
    const int t  = threadIdx.x;
    const int n  = blockIdx.x / NWIN, j = blockIdx.x % NWIN;
    const int qs = (3 * j + 80) % NSEG;

    float acc[32];
    #pragma unroll
    for (int i = 0; i < 32; ++i) acc[i] = 0.f;
    for (int tt = 0; tt < 20; ++tt) {
        const int q = (qs + tt) % NSEG;
        const float* p = wx + ((size_t)n * NSEG + q) * (KCL * CDIM);
        if (tt + 1 < 20) {   // prefetch next segment slice into GL2
            const int qn = (qs + tt + 1) % NSEG;
            __builtin_prefetch(wx + ((size_t)n * NSEG + qn) * (KCL * CDIM) + t, 0, 1);
        }
        #pragma unroll
        for (int i = 0; i < 32; ++i) acc[i] += p[t + 256 * i];
    }
    if (t < KCL) {
        float s = 0.f;
        const float* ap = aseg + ((size_t)n * KCL + t) * NSEG;
        for (int tt = 0; tt < 20; ++tt) s += ap[(qs + tt) % NSEG];
        sAw[t] = s;
    }
    __syncthreads();
    #pragma unroll
    for (int i = 0; i < 32; ++i) {
        const int kc = t + 256 * i;
        vals[kc] = acc[i] - centers[kc] * sAw[kc >> 7];
    }
    __syncthreads();
    if (t < KCL) {   // intra-cluster L2 norm
        float ss = 0.f;
        for (int c = 0; c < CDIM; ++c) { const float v = vals[(t << 7) + c]; ss += v * v; }
        sNorm[t] = 1.f / fmaxf(sqrtf(ss), 1e-12f);
    }
    __syncthreads();
    float lss = 0.f;
    #pragma unroll
    for (int i = 0; i < 32; ++i) {
        const int kc = t + 256 * i;
        const float v = vals[kc] * sNorm[kc >> 7];
        vals[kc] = v;
        lss += v * v;
    }
    red[t] = lss;
    __syncthreads();
    for (int off = 128; off > 0; off >>= 1) {
        if (t < off) red[t] += red[t + off];
        __syncthreads();
    }
    const float g = 1.f / fmaxf(sqrtf(red[0]), 1e-12f);
    _Float16* vp = vlad + (size_t)blockIdx.x * (KCL * CDIM);
    #pragma unroll
    for (int i = 0; i < 32; ++i) {
        const int kc = t + 256 * i;
        vp[kc] = (_Float16)(vals[kc] * g);
    }
}

// ---------------------------------------------------------------------------
// Kernel 4: MLP (240x8192 @ 8192x256, WMMA) + bias + per-row L2 norm -> out
// grid: 15 (row tiles of 16), block 256 (8 waves x 32 output cols)
// ---------------------------------------------------------------------------
__global__ __launch_bounds__(256) void k4_mlp(const _Float16* __restrict__ vlad,
                                              const float* __restrict__ mlp_w,
                                              const float* __restrict__ mlp_b,
                                              float* __restrict__ out) {
    __shared__ __align__(32) _Float16 sAf[TILE];          // A tile, 1 KB
    __shared__ __align__(32) _Float16 sBw[8][2 * TILE];   // per-wave B (2 col tiles), 16 KB
    __shared__ float sOut[16][OUTD];                      // 16 KB
    __shared__ float red[256];
    __shared__ float rn[16];
    const int t    = threadIdx.x;
    const int lane = t & 31, wave = t >> 5, hf = lane >> 4, l16 = lane & 15;
    const int row0 = blockIdx.x * 16;
    const int o0   = wave * 32;

    v8f acc0 = {}, acc1 = {};
    for (int ch = 0; ch < 256; ++ch) {      // K = 8192 = 256 chunks of 32
        const int kc0 = ch * 32;
        {   // cooperative A stage (2 elems/thread), fragment-swizzled
            int e = t, r = e >> 5, kk = e & 31;
            sAf[a_swz(r, kk)] = vlad[(size_t)(row0 + r) * 8192 + kc0 + kk];
            e = t + 256; r = e >> 5; kk = e & 31;
            sAf[a_swz(r, kk)] = vlad[(size_t)(row0 + r) * 8192 + kc0 + kk];
        }
        // per-wave B stage: k-row = lane, col tile = oi>>4, col = oi&15
        for (int oi = 0; oi < 32; ++oi)
            sBw[wave][(oi >> 4) * TILE + b_swz(lane, oi & 15)] =
                (_Float16)mlp_w[(size_t)(o0 + oi) * 8192 + kc0 + lane];
        if (ch + 1 < 256) {   // prefetch next K-chunk into GL2
            __builtin_prefetch(vlad + (size_t)(row0 + (t >> 5)) * 8192 + kc0 + 32, 0, 1);
            __builtin_prefetch(mlp_w + (size_t)(o0 + lane) * 8192 + kc0 + 32, 0, 1);
        }
        __syncthreads();
        const v16h af  = load_frag(sAf, lane);
        const v16h bf0 = load_frag(&sBw[wave][0],    lane);
        const v16h bf1 = load_frag(&sBw[wave][TILE], lane);
        acc0 = WMMA_F16(af, bf0, acc0);
        acc1 = WMMA_F16(af, bf1, acc1);
        __syncthreads();
    }
    #pragma unroll
    for (int r = 0; r < 8; ++r) {
        const int m = (hf << 3) + r;
        sOut[m][o0 + l16]      = acc0[r] + mlp_b[o0 + l16];
        sOut[m][o0 + 16 + l16] = acc1[r] + mlp_b[o0 + 16 + l16];
    }
    __syncthreads();
    const int r = t >> 4, off = t & 15;
    float ss = 0.f;
    #pragma unroll
    for (int i = 0; i < 16; ++i) { const float v = sOut[r][off + 16 * i]; ss += v * v; }
    red[t] = ss;
    __syncthreads();
    if (off == 0) {
        float s = 0.f;
        #pragma unroll
        for (int i = 0; i < 16; ++i) s += red[(r << 4) + i];
        rn[r] = 1.f / fmaxf(sqrtf(s), 1e-12f);
    }
    __syncthreads();
    float* op = out + (size_t)(row0 + r) * OUTD;
    #pragma unroll
    for (int i = 0; i < 16; ++i)
        op[off + 16 * i] = sOut[r][off + 16 * i] * rn[r];
}

// ---------------------------------------------------------------------------
extern "C" void kernel_launch(void* const* d_in, const int* in_sizes, int n_in,
                              void* d_out, int out_size, void* d_ws, size_t ws_size,
                              hipStream_t stream) {
    (void)in_sizes; (void)n_in; (void)out_size; (void)ws_size;
    const float* x  = (const float*)d_in[0];
    const float* cc = (const float*)d_in[1];   // cluster_centers (64x128 = kc flat)
    const float* cw = (const float*)d_in[2];
    const float* cb = (const float*)d_in[3];
    const float* mw = (const float*)d_in[4];
    const float* mb = (const float*)d_in[5];
    float* out = (float*)d_out;

    char* w = (char*)d_ws;
    size_t off = 0;
    auto alloc = [&](size_t bytes) -> void* {
        void* p = w + off;
        off += (bytes + 255) & ~(size_t)255;
        return p;
    };
    _Float16* a_f16 = (_Float16*)alloc((size_t)NB * KCL * HW * sizeof(_Float16));          // 14.7 MB
    float*    wxseg = (float*)   alloc((size_t)NB * NSEG * KCL * CDIM * sizeof(float));    // 23.6 MB
    float*    aseg  = (float*)   alloc((size_t)NB * KCL * NSEG * sizeof(float));           // 184 KB
    _Float16* vlad  = (_Float16*)alloc((size_t)NB * NWIN * KCL * CDIM * sizeof(_Float16)); // 3.9 MB

    k1_assign<<<dim3(NB * (HW / 64)), dim3(128), 0, stream>>>(x, cw, cb, a_f16);
    k2_wxseg<<<dim3(NB * NSEG), dim3(256), 0, stream>>>(x, a_f16, wxseg, aseg);
    k3_window_norm<<<dim3(NB * NWIN), dim3(256), 0, stream>>>(wxseg, aseg, cc, vlad);
    k4_mlp<<<dim3((NB * NWIN) / 16), dim3(256), 0, stream>>>(vlad, mw, mb, out);
}
